// VectorQuantizer2_19765439496685
// MI455X (gfx1250) — matrix operand order repaired and checked by reading, not compiled, not run on hardware
//
#include <hip/hip_runtime.h>

typedef __bf16 v16bf __attribute__((ext_vector_type(16)));
typedef float  v8f   __attribute__((ext_vector_type(8)));

union Pack {
    uint4 q[2];
    v16bf v;
};

// ---- constants ------------------------------------------------------------
#define B_     512
#define C_     32
#define H_     64
#define V_     4096
#define NELEM  (B_ * C_ * H_)   // 1048576

// round-to-nearest-even f32 -> bf16 (stored as ushort)
__device__ __forceinline__ unsigned short f2bf(float x) {
    unsigned int u = __float_as_uint(x);
    u += 0x7FFFu + ((u >> 16) & 1u);
    return (unsigned short)(u >> 16);
}

__device__ __forceinline__ float cubicw(float d) {
    d = fabsf(d);
    const float a = -0.75f;
    if (d <= 1.f) return ((a + 2.f) * d - (a + 3.f)) * d * d + 1.f;
    if (d < 2.f)  return ((a * d - 5.f * a) * d + 8.f * a) * d - 4.f * a;
    return 0.f;
}

// ---- init: f_rest = f, f_hat = 0 ------------------------------------------
__global__ __launch_bounds__(256) void initKernel(const float* __restrict__ f,
                                                  float* __restrict__ f_rest,
                                                  float* __restrict__ f_hat) {
    int e = blockIdx.x * 256 + threadIdx.x;
    if (e < NELEM) {
        f_rest[e] = f[e];
        f_hat[e]  = 0.f;
    }
}

// ---- prep: bf16 codebook + ||e||^2 ----------------------------------------
__global__ __launch_bounds__(256) void prepKernel(const float* __restrict__ emb,
                                                  unsigned short* __restrict__ emb_bf,
                                                  float* __restrict__ esq) {
    int e = blockIdx.x * 256 + threadIdx.x;
    if (e < V_ * C_) emb_bf[e] = f2bf(emb[e]);
    if (e < V_) {
        const float* r = emb + e * C_;
        float s = 0.f;
        #pragma unroll
        for (int i = 0; i < C_; ++i) s = fmaf(r[i], r[i], s);
        esq[e] = s;
    }
}

// ---- area down: f_rest (B,C,H) -> rest_bf ((B*pn), C) in bf16 -------------
__global__ __launch_bounds__(256) void areaDownKernel(const float* __restrict__ f_rest,
                                                      unsigned short* __restrict__ rest_bf,
                                                      int pn, int total) {
    int e = blockIdx.x * 256 + threadIdx.x;
    if (e >= total) return;
    int c = e & 31;
    int p = (e >> 5) % pn;
    int b = e / (32 * pn);
    int stride = H_ / pn;
    const float* src = f_rest + ((b * C_ + c) * H_) + p * stride;
    float s = 0.f;
    for (int i = 0; i < stride; ++i) s += src[i];
    rest_bf[e] = f2bf(s * (1.f / (float)stride));
}

// one 16-code tile: 2 WMMAs + argmin update for 32 rows
__device__ __forceinline__ void stepTile(const v16bf a0, const v16bf a1, const v16bf bv,
                                         float en, int code,
                                         float (&best0)[8], int (&bidx0)[8],
                                         float (&best1)[8], int (&bidx1)[8]) {
    v8f c0 = {0.f, 0.f, 0.f, 0.f, 0.f, 0.f, 0.f, 0.f};
    v8f c1 = {0.f, 0.f, 0.f, 0.f, 0.f, 0.f, 0.f, 0.f};
    c0 = __builtin_amdgcn_wmma_f32_16x16x32_bf16(false, a0, false, bv,
                                                 (short)0, c0, false, false);
    c1 = __builtin_amdgcn_wmma_f32_16x16x32_bf16(false, a1, false, bv,
                                                 (short)0, c1, false, false);
    // slot r holds (M = r + 8*hi, N = nl): d' = ||e||^2 - 2 r.e
    #pragma unroll
    for (int r = 0; r < 8; ++r) {
        float d0 = fmaf(-2.f, c0[r], en);
        float d1 = fmaf(-2.f, c1[r], en);
        if (d0 < best0[r]) { best0[r] = d0; bidx0[r] = code; }
        if (d1 < best1[r]) { best1[r] = d1; bidx1[r] = code; }
    }
}

// ---- code search: 32 rows/block, 8 waves x 512 codes, 2x WMMA bf16 K=32 ---
// B tiles and ||e||^2 are ping-pong double-buffered (no rotation movs).
__global__ __launch_bounds__(256) void argminKernel(const unsigned short* __restrict__ rest_bf,
                                                    const unsigned short* __restrict__ emb_bf,
                                                    const float* __restrict__ esq,
                                                    int* __restrict__ idx_out) {
    __shared__ float s_bd[8][32];
    __shared__ int   s_bi[8][32];

    const int tid  = threadIdx.x;
    const int wave = tid >> 5;
    const int lane = tid & 31;
    const int hi   = lane >> 4;     // which half of channels / M offset
    const int nl   = lane & 15;     // N index inside tile / row index for A
    const int row0 = blockIdx.x * 32 + nl;
    const int row1 = row0 + 16;

    // A operands: 16x32 bf16; lane(hi,nl) holds row nl, channels {hi*8..}+{16+hi*8..}
    Pack pa0, pa1;
    pa0.q[0] = *(const uint4*)(rest_bf + row0 * C_ + hi * 8);
    pa0.q[1] = *(const uint4*)(rest_bf + row0 * C_ + 16 + hi * 8);
    pa1.q[0] = *(const uint4*)(rest_bf + row1 * C_ + hi * 8);
    pa1.q[1] = *(const uint4*)(rest_bf + row1 * C_ + 16 + hi * 8);
    const v16bf a0 = pa0.v;
    const v16bf a1 = pa1.v;

    float best0[8], best1[8];
    int   bidx0[8], bidx1[8];
    #pragma unroll
    for (int r = 0; r < 8; ++r) {
        best0[r] = 3.4e38f; bidx0[r] = 0x7fffffff;
        best1[r] = 3.4e38f; bidx1[r] = 0x7fffffff;
    }

    const int nstart = wave * 512;
    const unsigned short* bbase = emb_bf + (nstart + nl) * C_ + hi * 8;

    // prefetch tile 0 into buffer 0
    Pack b0, b1;
    float en0, en1;
    b0.q[0] = *(const uint4*)(bbase);
    b0.q[1] = *(const uint4*)(bbase + 16);
    en0 = esq[nstart + nl];

    for (int tt = 0; tt < 16; ++tt) {
        const int t0 = tt * 2;
        // prefetch tile t0+1 into buffer 1, compute tile t0 from buffer 0
        b1.q[0] = *(const uint4*)(bbase + (t0 + 1) * 16 * C_);
        b1.q[1] = *(const uint4*)(bbase + (t0 + 1) * 16 * C_ + 16);
        en1 = esq[nstart + (t0 + 1) * 16 + nl];
        stepTile(a0, a1, b0.v, en0, nstart + t0 * 16 + nl,
                 best0, bidx0, best1, bidx1);

        // prefetch tile t0+2 into buffer 0, compute tile t0+1 from buffer 1
        // (final iteration's prefetch reads a few bytes past emb_bf/esq; the
        //  ws layout keeps this inside d_ws and the values are never used)
        b0.q[0] = *(const uint4*)(bbase + (t0 + 2) * 16 * C_);
        b0.q[1] = *(const uint4*)(bbase + (t0 + 2) * 16 * C_ + 16);
        en0 = esq[nstart + (t0 + 2) * 16 + nl];
        stepTile(a0, a1, b1.v, en1, nstart + (t0 + 1) * 16 + nl,
                 best0, bidx0, best1, bidx1);
    }

    // reduce over N (16 lanes within each half); tie -> smaller index
    #pragma unroll
    for (int r = 0; r < 8; ++r) {
        float d = best0[r];
        int   i = bidx0[r];
        #pragma unroll
        for (int m = 1; m < 16; m <<= 1) {
            float od = __shfl_xor(d, m, 32);
            int   oi = __shfl_xor(i, m, 32);
            if (od < d || (od == d && oi < i)) { d = od; i = oi; }
        }
        if (nl == 0) { s_bd[wave][r + hi * 8] = d; s_bi[wave][r + hi * 8] = i; }

        float e1 = best1[r];
        int   i1 = bidx1[r];
        #pragma unroll
        for (int m = 1; m < 16; m <<= 1) {
            float od = __shfl_xor(e1, m, 32);
            int   oi = __shfl_xor(i1, m, 32);
            if (od < e1 || (od == e1 && oi < i1)) { e1 = od; i1 = oi; }
        }
        if (nl == 0) { s_bd[wave][16 + r + hi * 8] = e1; s_bi[wave][16 + r + hi * 8] = i1; }
    }
    __syncthreads();

    if (tid < 32) {
        float d = s_bd[0][tid];
        int   i = s_bi[0][tid];
        for (int w = 1; w < 8; ++w) {
            float od = s_bd[w][tid];
            int   oi = s_bi[w][tid];
            if (od < d || (od == d && oi < i)) { d = od; i = oi; }
        }
        idx_out[blockIdx.x * 32 + tid] = i;
    }
}

// ---- gather + bicubic up + 3-tap conv + blend + state update --------------
// Thread mapping: co = lane (conflict-free [ci][k][co] weight reads),
// 8 consecutive y per thread (register-cached h_up window, broadcast reads).
__global__ __launch_bounds__(256) void upPhiKernel(const int* __restrict__ idx,
                                                   const float* __restrict__ emb,
                                                   const float* __restrict__ phi_w,
                                                   const float* __restrict__ phi_b,
                                                   const float* __restrict__ f,
                                                   float* __restrict__ f_rest,
                                                   float* __restrict__ f_hat,
                                                   float* __restrict__ partial,
                                                   int pn, int pi) {
    __shared__ float s_hq[H_ * C_];      // (p, c) quantized latents
    __shared__ float s_up[C_][H_ + 2];   // upsampled, halo for conv
    __shared__ float s_w[C_][3][C_];     // conv weights, [ci][kh][co]
    __shared__ float s_b[C_];
    __shared__ float s_wy[H_][4];
    __shared__ int   s_i0[H_];
    __shared__ float s_red[8];

    const int b   = blockIdx.x;
    const int tid = threadIdx.x;

    for (int e = tid; e < C_ * C_ * 3; e += 256) {
        int kh = e % 3, ci = (e / 3) & 31, co = e / 96;
        s_w[ci][kh][co] = phi_w[(((pi * C_ + co) * C_ + ci) * 3 + kh) * 3 + 1];
    }
    if (tid < C_) s_b[tid] = phi_b[pi * C_ + tid];

    for (int e = tid; e < pn * C_; e += 256) {
        int q = idx[b * pn + (e >> 5)];
        s_hq[e] = emb[q * C_ + (e & 31)];
    }

    if (tid < H_) {
        float scale = (float)pn / (float)H_;
        float src = ((float)tid + 0.5f) * scale - 0.5f;
        float i0f = floorf(src);
        float t = src - i0f;
        s_i0[tid]    = (int)i0f;
        s_wy[tid][0] = cubicw(t + 1.f);
        s_wy[tid][1] = cubicw(t);
        s_wy[tid][2] = cubicw(1.f - t);
        s_wy[tid][3] = cubicw(2.f - t);
    }
    __syncthreads();

    for (int e = tid; e < C_ * H_; e += 256) {
        int y = e & 63, c = e >> 6;
        int i0 = s_i0[y];
        float acc = 0.f;
        #pragma unroll
        for (int k = 0; k < 4; ++k) {
            int p = min(max(i0 - 1 + k, 0), pn - 1);
            acc = fmaf(s_wy[y][k], s_hq[p * C_ + c], acc);
        }
        s_up[c][y + 1] = acc;
    }
    if (tid < C_) { s_up[tid][0] = 0.f; s_up[tid][H_ + 1] = 0.f; }
    __syncthreads();

    const int co = tid & 31;
    const int yb = (tid >> 5) * 8;     // 8 waves x 8 y = 64

    float acc[8];
    #pragma unroll
    for (int j = 0; j < 8; ++j) acc[j] = s_b[co];

    for (int ci = 0; ci < C_; ++ci) {
        float u[10];
        #pragma unroll
        for (int j = 0; j < 10; ++j) u[j] = s_up[ci][yb + j];   // broadcast
        float w0 = s_w[ci][0][co];
        float w1 = s_w[ci][1][co];
        float w2 = s_w[ci][2][co];
        #pragma unroll
        for (int j = 0; j < 8; ++j)
            acc[j] = fmaf(w0, u[j], fmaf(w1, u[j + 1], fmaf(w2, u[j + 2], acc[j])));
    }

    float lsq = 0.f;
    #pragma unroll
    for (int j = 0; j < 8; ++j) {
        int y = yb + j;
        float hf = 0.5f * s_up[co][y + 1] + 0.5f * acc[j];   // QRESI = 0.5
        int o = (b * C_ + co) * H_ + y;
        float fh = f_hat[o] + hf;
        f_hat[o]   = fh;
        f_rest[o] -= hf;
        float d = fh - f[o];
        lsq = fmaf(d, d, lsq);
    }

    // deterministic block reduction
    #pragma unroll
    for (int m = 1; m < 32; m <<= 1) lsq += __shfl_xor(lsq, m, 32);
    if ((tid & 31) == 0) s_red[tid >> 5] = lsq;
    __syncthreads();
    if (tid == 0) {
        float s = 0.f;
        #pragma unroll
        for (int w = 0; w < 8; ++w) s += s_red[w];
        partial[b] = s;
    }
}

// ---- final loss: fixed-order sum of 7*512 partials ------------------------
__global__ void lossKernel(const float* __restrict__ partials, float* __restrict__ out_loss) {
    if (threadIdx.x == 0 && blockIdx.x == 0) {
        float s = 0.f;
        for (int i = 0; i < 7 * B_; ++i) s += partials[i];
        // (1 + BETA) / (SN * numel)
        *out_loss = s * (1.25f / (7.f * (float)NELEM));
    }
}

extern "C" void kernel_launch(void* const* d_in, const int* in_sizes, int n_in,
                              void* d_out, int out_size, void* d_ws, size_t ws_size,
                              hipStream_t stream) {
    const float* f     = (const float*)d_in[0];
    const float* emb   = (const float*)d_in[1];
    const float* phi_w = (const float*)d_in[2];
    const float* phi_b = (const float*)d_in[3];

    float* f_hat    = (float*)d_out;          // NELEM floats
    float* loss_out = f_hat + NELEM;          // 1 float

    char* ws = (char*)d_ws;
    float*          f_rest   = (float*)(ws + 0);                  // 4 MB
    unsigned short* rest_bf  = (unsigned short*)(ws + 4194304);   // 2 MB (max 32768 rows x 32)
    unsigned short* emb_bf   = (unsigned short*)(ws + 6291456);   // 256 KB (+overrun pad below)
    float*          esq      = (float*)(ws + 6553600);            // 16 KB  (+overrun pad below)
    int*            idxbuf   = (int*)(ws + 6569984);              // 128 KB
    float*          partials = (float*)(ws + 6701056);            // 7*512 floats
    // prefetch overrun from argminKernel's last iteration lands in the
    // adjacent ws regions above (valid device memory, values unused).

    initKernel<<<NELEM / 256, 256, 0, stream>>>(f, f_rest, f_hat);
    prepKernel<<<(V_ * C_) / 256, 256, 0, stream>>>(emb, emb_bf, esq);

    const int pns[7] = {1, 2, 4, 8, 16, 32, 64};
    const int pis[7] = {0, 0, 1, 1, 2, 3, 3};   // argmin(|ticks - si/6|)

    for (int si = 0; si < 7; ++si) {
        int pn = pns[si];
        int total = B_ * pn * C_;
        areaDownKernel<<<(total + 255) / 256, 256, 0, stream>>>(f_rest, rest_bf, pn, total);
        argminKernel<<<(B_ * pn) / 32, 256, 0, stream>>>(rest_bf, emb_bf, esq, idxbuf);
        upPhiKernel<<<B_, 256, 0, stream>>>(idxbuf, emb, phi_w, phi_b, f, f_rest, f_hat,
                                            partials + si * B_, pn, pis[si]);
    }
    lossKernel<<<1, 32, 0, stream>>>(partials, loss_out);
}
